// Task2Net_79362405695742
// MI455X (gfx1250) — matrix operand order repaired and checked by reading, not compiled, not run on hardware
//
#include <hip/hip_runtime.h>
#include <hip/hip_bf16.h>

typedef __attribute__((ext_vector_type(16))) _Float16 v16h;
typedef __attribute__((ext_vector_type(8)))  float    v8f;
typedef __attribute__((ext_vector_type(2)))  float    v2f;   // clang vector for NT loads

#define N_NODES 100000
#define EMB     64
#define NHEAD   3
#define HID     8
#define D1      24      // NHEAD*HID
#define OUTD    16
#define BQ      1024

// ---------- monotone float<->uint encoding for atomicMax on floats ----------
static __device__ __forceinline__ unsigned encf(float x) {
    unsigned u = __float_as_uint(x);
    return (u & 0x80000000u) ? ~u : (u | 0x80000000u);
}
static __device__ __forceinline__ float decf(unsigned e) {
    unsigned u = (e & 0x80000000u) ? (e & 0x7FFFFFFFu) : ~e;
    return __uint_as_float(u);
}
static __device__ __forceinline__ float lrelu(float x) { return x > 0.f ? x : 0.01f * x; }
static __device__ __forceinline__ float eluf(float x)  { return x > 0.f ? x : (__expf(x) - 1.f); }

// ---------------- zero-init of atomic accumulator region ----------------
__global__ void k_zero(float* w, size_t n) {
    size_t i = (size_t)blockIdx.x * blockDim.x + threadIdx.x;
    size_t stride = (size_t)gridDim.x * blockDim.x;
    for (; i < n; i += stride) w[i] = 0.f;
}

// ---------------- layer 1 node GEMM: z1 = emb @ W1, plus s1/d1 scalars ----------------
__global__ void k_z1(const float* __restrict__ emb, const float* __restrict__ W1,
                     const float* __restrict__ a1,
                     float* __restrict__ z1, float* __restrict__ s1, float* __restrict__ d1) {
    __shared__ float Wl[EMB * D1];   // Wl[k*24 + h*8 + o]
    __shared__ float al[NHEAD * 16];
    int tid = threadIdx.x;
    for (int idx = tid; idx < NHEAD * EMB * HID; idx += blockDim.x) {
        int h = idx / (EMB * HID);
        int rem = idx % (EMB * HID);
        int k = rem / HID, o = rem % HID;
        Wl[k * D1 + h * HID + o] = W1[idx];
    }
    if (tid < NHEAD * 16) al[tid] = a1[tid];
    __syncthreads();

    int n = blockIdx.x * blockDim.x + tid;
    if (n >= N_NODES) return;
    float acc[D1];
#pragma unroll
    for (int j = 0; j < D1; ++j) acc[j] = 0.f;
    const float4* e4 = (const float4*)(emb + (size_t)n * EMB);
#pragma unroll 4
    for (int q = 0; q < EMB / 4; ++q) {
        float4 ev = e4[q];
        float ex[4] = {ev.x, ev.y, ev.z, ev.w};
#pragma unroll
        for (int t = 0; t < 4; ++t) {
            int k = q * 4 + t;
#pragma unroll
            for (int j = 0; j < D1; ++j) acc[j] += ex[t] * Wl[k * D1 + j];
        }
    }
#pragma unroll
    for (int j = 0; j < D1; ++j) z1[(size_t)n * D1 + j] = acc[j];
#pragma unroll
    for (int h = 0; h < NHEAD; ++h) {
        float sv = 0.f, dv = 0.f;
#pragma unroll
        for (int o = 0; o < HID; ++o) {
            float zz = acc[h * HID + o];
            sv += zz * al[h * 16 + o];
            dv += zz * al[h * 16 + 8 + o];
        }
        s1[n * NHEAD + h] = sv;
        d1[n * NHEAD + h] = dv;
    }
}

// ---------------- layer 1 edge pass A: segment max ----------------
__global__ void k_emax1(const int* __restrict__ src, const int* __restrict__ dst,
                        const float* __restrict__ s1, const float* __restrict__ d1,
                        unsigned* __restrict__ m1, int E) {
    int i = blockIdx.x * blockDim.x + threadIdx.x;
    if (i >= E) return;
    int s = src[i], t = dst[i];
#pragma unroll
    for (int h = 0; h < NHEAD; ++h) {
        float e = lrelu(s1[s * NHEAD + h] + d1[t * NHEAD + h]);
        atomicMax(&m1[t * NHEAD + h], encf(e));
    }
}

// ---------------- layer 1 edge pass B: softmax-weighted scatter-sum ----------------
__global__ void k_esum1(const int* __restrict__ src, const int* __restrict__ dst,
                        const float* __restrict__ s1, const float* __restrict__ d1,
                        const unsigned* __restrict__ m1, const float* __restrict__ z1,
                        float* __restrict__ den1, float* __restrict__ num1, int E) {
    int i = blockIdx.x * blockDim.x + threadIdx.x;
    if (i >= E) return;
    int s = src[i], t = dst[i];
#pragma unroll
    for (int h = 0; h < NHEAD; ++h) {
        float e = lrelu(s1[s * NHEAD + h] + d1[t * NHEAD + h]);
        float alpha = __expf(e - decf(m1[t * NHEAD + h]));
        atomicAdd(&den1[t * NHEAD + h], alpha);
#pragma unroll
        for (int o = 0; o < HID; ++o)
            atomicAdd(&num1[(size_t)t * D1 + h * HID + o],
                      alpha * z1[(size_t)s * D1 + h * HID + o]);
    }
}

// ---------------- h1 = elu(num1/den1); z2 = h1 @ W2; s2/d2 scalars ----------------
__global__ void k_h1z2(const float* __restrict__ num1, const float* __restrict__ den1,
                       const float* __restrict__ W2, const float* __restrict__ a2,
                       float* __restrict__ z2, float* __restrict__ s2, float* __restrict__ d2) {
    __shared__ float Wl[D1 * OUTD];
    __shared__ float al[2 * OUTD];
    int tid = threadIdx.x;
    for (int idx = tid; idx < D1 * OUTD; idx += blockDim.x) Wl[idx] = W2[idx];
    if (tid < 2 * OUTD) al[tid] = a2[tid];
    __syncthreads();

    int n = blockIdx.x * blockDim.x + tid;
    if (n >= N_NODES) return;
    float h1v[D1];
#pragma unroll
    for (int j = 0; j < D1; ++j)
        h1v[j] = eluf(num1[(size_t)n * D1 + j] / den1[n * NHEAD + j / HID]);
    float za[OUTD];
#pragma unroll
    for (int c = 0; c < OUTD; ++c) za[c] = 0.f;
#pragma unroll 4
    for (int j = 0; j < D1; ++j) {
        float hv = h1v[j];
#pragma unroll
        for (int c = 0; c < OUTD; ++c) za[c] += hv * Wl[j * OUTD + c];
    }
    float sv = 0.f, dv = 0.f;
#pragma unroll
    for (int c = 0; c < OUTD; ++c) {
        z2[(size_t)n * OUTD + c] = za[c];
        sv += za[c] * al[c];
        dv += za[c] * al[OUTD + c];
    }
    s2[n] = sv;
    d2[n] = dv;
}

// ---------------- layer 2 edge passes (1 head, o=16) ----------------
__global__ void k_emax2(const int* __restrict__ src, const int* __restrict__ dst,
                        const float* __restrict__ s2, const float* __restrict__ d2,
                        unsigned* __restrict__ m2, int E) {
    int i = blockIdx.x * blockDim.x + threadIdx.x;
    if (i >= E) return;
    float e = lrelu(s2[src[i]] + d2[dst[i]]);
    atomicMax(&m2[dst[i]], encf(e));
}

__global__ void k_esum2(const int* __restrict__ src, const int* __restrict__ dst,
                        const float* __restrict__ s2, const float* __restrict__ d2,
                        const unsigned* __restrict__ m2, const float* __restrict__ z2,
                        float* __restrict__ den2, float* __restrict__ num2, int E) {
    int i = blockIdx.x * blockDim.x + threadIdx.x;
    if (i >= E) return;
    int s = src[i], t = dst[i];
    float e = lrelu(s2[s] + d2[t]);
    float alpha = __expf(e - decf(m2[t]));
    atomicAdd(&den2[t], alpha);
#pragma unroll
    for (int c = 0; c < OUTD; ++c)
        atomicAdd(&num2[(size_t)t * OUTD + c], alpha * z2[(size_t)s * OUTD + c]);
}

// ---------------- item_embeds = num2/den2 ----------------
__global__ void k_item(const float* __restrict__ num2, const float* __restrict__ den2,
                       float* __restrict__ item) {
    int idx = blockIdx.x * blockDim.x + threadIdx.x;
    if (idx >= N_NODES * OUTD) return;
    item[idx] = num2[idx] / den2[idx / OUTD];
}

// ---------------- WMMA query pooling: qsum += Q @ item, qcnt += rowsum(Q) ----------------
// A (16x32 f16): lane L holds row M=L%16; half=L/16 selects K {0-7,16-23} vs {8-15,24-31}
// B (32x16 f16): lane L holds col N=L%16; half=L/16 selects K {0-7,16-23} vs {8-15,24-31} (same as A)
// C (16x16 f32): VGPR r -> M = r + 8*half, N = L%16
static __device__ __forceinline__ void load_frags(const float* __restrict__ q,
                                                  const float* __restrict__ item,
                                                  int k0, int row, int hf, int col,
                                                  v16h& a, v16h& b, float& cnt) {
#pragma unroll
    for (int v = 0; v < 8; ++v) {
        int ka = k0 + ((v < 4) ? 0 : 16) + hf * 8 + (v & 3) * 2;
        // streaming read of 410MB queries: non-temporal so item (6.4MB) stays in L2
        const v2f qv = __builtin_nontemporal_load(
            (const v2f*)(q + (size_t)row * N_NODES + ka));
        a[2 * v]     = (_Float16)qv.x;
        a[2 * v + 1] = (_Float16)qv.y;
        cnt += qv.x + qv.y;
    }
#pragma unroll
    for (int e = 0; e < 16; ++e) {
        int kb = k0 + ((e < 8) ? 0 : 16) + hf * 8 + (e & 7);
        b[e] = (_Float16)item[(size_t)kb * OUTD + col];
    }
}

__global__ void k_qgemm(const float* __restrict__ q, const float* __restrict__ item,
                        float* __restrict__ qsum, float* __restrict__ qcnt) {
    const int lane = threadIdx.x & 31;
    const int wave = threadIdx.x >> 5;
    const int rowBase = blockIdx.x * 16;
    const int row = rowBase + (lane & 15);
    const int hf = lane >> 4;
    const int col = lane & 15;
    const int nChunks = N_NODES / 32;           // 3125 (exact)
    const int stride = gridDim.y * 8;

    v8f c0 = {}, c1 = {};
    float cnt = 0.f;
    int ch = blockIdx.y * 8 + wave;
    // pairwise: two independent accumulator chains hide WMMA pipeline latency
    for (; ch + stride < nChunks; ch += 2 * stride) {
        int chn = ch + 2 * stride;
        if (chn < nChunks)
            __builtin_prefetch(q + (size_t)row * N_NODES + chn * 32, 0, 0);
        v16h a0, b0, a1v, b1v;
        load_frags(q, item, ch * 32, row, hf, col, a0, b0, cnt);
        load_frags(q, item, (ch + stride) * 32, row, hf, col, a1v, b1v, cnt);
        c0 = __builtin_amdgcn_wmma_f32_16x16x32_f16(false, a0, false, b0,
                                                    (short)0, c0, false, false);
        c1 = __builtin_amdgcn_wmma_f32_16x16x32_f16(false, a1v, false, b1v,
                                                    (short)0, c1, false, false);
    }
    if (ch < nChunks) {
        v16h a0, b0;
        load_frags(q, item, ch * 32, row, hf, col, a0, b0, cnt);
        c0 = __builtin_amdgcn_wmma_f32_16x16x32_f16(false, a0, false, b0,
                                                    (short)0, c0, false, false);
    }
#pragma unroll
    for (int r = 0; r < 8; ++r)
        atomicAdd(&qsum[(size_t)(rowBase + r + 8 * hf) * OUTD + col], c0[r] + c1[r]);
    atomicAdd(&qcnt[row], cnt);
}

// ---------------- finalize: mean-divide + pos/neg gathers ----------------
__global__ void k_final(const float* __restrict__ qsum, const float* __restrict__ qcnt,
                        const float* __restrict__ item,
                        const int* __restrict__ pos, const int* __restrict__ neg,
                        float* __restrict__ out) {
    int idx = blockIdx.x * blockDim.x + threadIdx.x;
    if (idx >= BQ * OUTD) return;
    int bq = idx / OUTD, cc = idx % OUTD;
    out[idx] = qsum[idx] / qcnt[bq];
    out[BQ * OUTD + idx]     = item[(size_t)pos[bq] * OUTD + cc];
    out[2 * BQ * OUTD + idx] = item[(size_t)neg[bq] * OUTD + cc];
}

extern "C" void kernel_launch(void* const* d_in, const int* in_sizes, int n_in,
                              void* d_out, int out_size, void* d_ws, size_t ws_size,
                              hipStream_t stream) {
    const float* queries = (const float*)d_in[0];
    const int*   pos     = (const int*)d_in[1];
    const int*   neg     = (const int*)d_in[2];
    const float* emb     = (const float*)d_in[3];
    const float* W1      = (const float*)d_in[4];
    const float* a1      = (const float*)d_in[5];
    const float* W2      = (const float*)d_in[6];
    const float* a2      = (const float*)d_in[7];
    const int*   src     = (const int*)d_in[8];
    const int*   dst     = (const int*)d_in[9];
    const int    E       = in_sizes[8];
    float* out = (float*)d_out;

    // workspace carve (floats)
    float* w = (float*)d_ws;
    size_t o = 0;
    unsigned* m1 = (unsigned*)(w + o); o += (size_t)N_NODES * NHEAD;   // zero-init region start
    float* den1 = w + o;               o += (size_t)N_NODES * NHEAD;
    float* num1 = w + o;               o += (size_t)N_NODES * D1;
    unsigned* m2 = (unsigned*)(w + o); o += (size_t)N_NODES;
    float* den2 = w + o;               o += (size_t)N_NODES;
    float* num2 = w + o;               o += (size_t)N_NODES * OUTD;
    float* qsum = w + o;               o += (size_t)BQ * OUTD;
    float* qcnt = w + o;               o += (size_t)BQ;
    const size_t zeroN = o;                                            // zero-init region end
    float* z1 = w + o;                 o += (size_t)N_NODES * D1;
    float* s1 = w + o;                 o += (size_t)N_NODES * NHEAD;
    float* d1 = w + o;                 o += (size_t)N_NODES * NHEAD;
    float* z2 = w + o;                 o += (size_t)N_NODES * OUTD;
    float* s2 = w + o;                 o += (size_t)N_NODES;
    float* d2 = w + o;                 o += (size_t)N_NODES;
    float* item = w + o;               o += (size_t)N_NODES * OUTD;
    (void)ws_size; (void)n_in; (void)out_size;

    const int nodeBlocks = (N_NODES + 255) / 256;
    const int edgeBlocks = (E + 255) / 256;

    k_zero<<<2048, 256, 0, stream>>>(w, zeroN);
    k_z1<<<nodeBlocks, 256, 0, stream>>>(emb, W1, a1, z1, s1, d1);
    k_emax1<<<edgeBlocks, 256, 0, stream>>>(src, dst, s1, d1, m1, E);
    k_esum1<<<edgeBlocks, 256, 0, stream>>>(src, dst, s1, d1, m1, z1, den1, num1, E);
    k_h1z2<<<nodeBlocks, 256, 0, stream>>>(num1, den1, W2, a2, z2, s2, d2);
    k_emax2<<<edgeBlocks, 256, 0, stream>>>(src, dst, s2, d2, m2, E);
    k_esum2<<<edgeBlocks, 256, 0, stream>>>(src, dst, s2, d2, m2, z2, den2, num2, E);
    k_item<<<(N_NODES * OUTD + 255) / 256, 256, 0, stream>>>(num2, den2, item);
    k_qgemm<<<dim3(BQ / 16, 24), 256, 0, stream>>>(queries, item, qsum, qcnt);
    k_final<<<(BQ * OUTD + 255) / 256, 256, 0, stream>>>(qsum, qcnt, item, pos, neg, out);
}
